// LinearAttention_32083405701223
// MI455X (gfx1250) — matrix operand order repaired and checked by reading, not compiled, not run on hardware
//
#include <hip/hip_runtime.h>
#include <cstdint>
#include <cstddef>

// ---------- CDNA5 WMMA types ----------
typedef __attribute__((ext_vector_type(16))) _Float16 v16h;
typedef __attribute__((ext_vector_type(8)))  float    v8f;
typedef __attribute__((ext_vector_type(4)))  unsigned int u32x4;
typedef __attribute__((ext_vector_type(4)))  float    f32x4;

union AFrag { u32x4 q2[2]; v16h v; };
union BFrag { unsigned int u[8]; v16h v; };
union H2    { _Float16 h[2]; unsigned int u; };

__device__ __forceinline__ v8f wmma_f16(v16h a, v16h b, v8f c) {
  // D = A(16x32) * B(32x16) + C, f32 accumulate
  return __builtin_amdgcn_wmma_f32_16x16x32_f16(
      /*neg_a=*/false, a, /*neg_b=*/false, b,
      /*c_mod=*/(short)0, c, /*reuse_a=*/false, /*reuse_b=*/false);
}

#define NTOT 16384   // 128*128 spatial
#define CDIM 256
#define QKVROWS 768
#define EPSV 1e-5f
#define NCHUNK 8     // context n-split for occupancy (deterministic partials)

// ---------------- Kernel 1: per-(b,c) spatial mean / rstd ----------------
__global__ __launch_bounds__(256) void la_stats_kernel(
    const float* __restrict__ x, float* __restrict__ meanX, float* __restrict__ rstdX) {
  __shared__ float s1[256], s2[256];
  const int row = blockIdx.x;          // b*256 + c
  const int t = threadIdx.x;
  const f32x4* p = (const f32x4*)(x + (size_t)row * NTOT);
  float s = 0.f, q = 0.f;
  for (int i = t; i < NTOT / 4; i += 256) {
    f32x4 v = p[i];
    s += v.x + v.y + v.z + v.w;
    q += v.x * v.x + v.y * v.y + v.z * v.z + v.w * v.w;
  }
  s1[t] = s; s2[t] = q; __syncthreads();
  for (int st = 128; st > 0; st >>= 1) {
    if (t < st) { s1[t] += s1[t + st]; s2[t] += s2[t + st]; }
    __syncthreads();
  }
  if (t == 0) {
    float mu = s1[0] / (float)NTOT;
    float var = s2[0] / (float)NTOT - mu * mu;
    meanX[row] = mu;
    rstdX[row] = rsqrtf(var + EPSV);
  }
}

// ---------------- Kernel 2: f32 -> f16 weight convert ----------------
__global__ __launch_bounds__(256) void la_cvt_kernel(
    const float* __restrict__ src, _Float16* __restrict__ dst, int n) {
  int i = blockIdx.x * 256 + threadIdx.x;
  if (i < n) dst[i] = (_Float16)src[i];
}

// ---------------- Kernel 3: QKV GEMM (WMMA f16), f16 output ----------------
// qkv16[b, 0..767, n] = f16( w_qkv[768,256] @ xn[256, n] ), xn normalized on the fly.
__global__ __launch_bounds__(256) void la_gemm_qkv_kernel(
    const float* __restrict__ x, const float* __restrict__ meanX,
    const float* __restrict__ rstdX, const _Float16* __restrict__ wq,
    _Float16* __restrict__ qkv) {
  __shared__ unsigned int ldsB[16 * 128];   // packed f16 K-pairs: [k/2][n]
  const int t = threadIdx.x;
  const int lane = t & 31;
  const int w = t >> 5;
  const int wm = w >> 2, wn = w & 3;        // 2x4 wave grid -> wave tile 64x32
  const int b = blockIdx.z;
  const int m0 = blockIdx.y * 128;
  const int n0 = blockIdx.x * 128;
  const int l15 = lane & 15;
  const int khalf = lane >> 4;

  v8f acc[4][2];
#pragma unroll
  for (int i = 0; i < 4; i++)
#pragma unroll
    for (int j = 0; j < 2; j++) acc[i][j] = {};

  for (int k0 = 0; k0 < CDIM; k0 += 32) {
    // stage normalized x tile [32k x 128n] as f16 pairs
#pragma unroll
    for (int i = 0; i < 8; i++) {
      int idx = i * 256 + t;                // 0..2047
      int k2 = idx >> 7;                    // 0..15
      int n = idx & 127;
      int c0 = k0 + 2 * k2;
      float v0 = x[((size_t)(b * CDIM + c0)) * NTOT + n0 + n];
      float v1 = x[((size_t)(b * CDIM + c0 + 1)) * NTOT + n0 + n];
      v0 = (v0 - meanX[b * CDIM + c0]) * rstdX[b * CDIM + c0];
      v1 = (v1 - meanX[b * CDIM + c0 + 1]) * rstdX[b * CDIM + c0 + 1];
      H2 p; p.h[0] = (_Float16)v0; p.h[1] = (_Float16)v1;
      ldsB[k2 * 128 + n] = p.u;
      // prefetch next K-slab of x into near cache (global_prefetch_b8)
      if (k0 + 32 < CDIM)
        __builtin_prefetch(&x[((size_t)(b * CDIM + c0 + 32 + (i & 1))) * NTOT + n0 + n], 0, 3);
    }
    __syncthreads();

    AFrag a[4];
#pragma unroll
    for (int mi = 0; mi < 4; mi++) {
      int m = m0 + wm * 64 + mi * 16 + l15;
      const _Float16* wr = wq + (size_t)m * CDIM + k0 + khalf * 16;
      a[mi].q2[0] = *(const u32x4*)wr;          // K +0..7 (lanes<16) / +16..23
      a[mi].q2[1] = *(const u32x4*)(wr + 8);    // K +8..15 / +24..31
    }
    BFrag bf[2];
#pragma unroll
    for (int nj = 0; nj < 2; nj++) {
      int n = wn * 32 + nj * 16 + l15;
      int kb = khalf * 8;                       // pair-row base
#pragma unroll
      for (int j = 0; j < 8; j++) bf[nj].u[j] = ldsB[(kb + j) * 128 + n];
    }
#pragma unroll
    for (int mi = 0; mi < 4; mi++)
#pragma unroll
      for (int nj = 0; nj < 2; nj++)
        acc[mi][nj] = wmma_f16(a[mi].v, bf[nj].v, acc[mi][nj]);
    __syncthreads();
  }

#pragma unroll
  for (int mi = 0; mi < 4; mi++)
#pragma unroll
    for (int nj = 0; nj < 2; nj++) {
      int mbase = m0 + wm * 64 + mi * 16 + (khalf ? 8 : 0);
      int n = n0 + wn * 32 + nj * 16 + l15;
#pragma unroll
      for (int v = 0; v < 8; v++) {
        int m = mbase + v;
        qkv[((size_t)(b * QKVROWS + m)) * NTOT + n] = (_Float16)acc[mi][nj][v];
      }
    }
}

// ------- Kernel 4: one-pass online k-softmax stats (max, sum-exp incl mem kv) --
__global__ __launch_bounds__(256) void la_kstats_kernel(
    const _Float16* __restrict__ qkv, const float* __restrict__ memkv,
    float* __restrict__ kmax, float* __restrict__ ksum) {
  __shared__ float red_m[256], red_s[256];
  const int idx = blockIdx.x;        // b*256 + h*32 + d
  const int b = idx >> 8, r = idx & 255;
  const int h = r >> 5, d = r & 31;
  const _Float16* row = qkv + ((size_t)(b * QKVROWS + 256 + r)) * NTOT;
  const unsigned int* row32 = (const unsigned int*)row;   // 8192 f16-pairs
  const float* memk = memkv + (h * 32 + d) * 4;           // mem_kv[0][h][d][m]
  const int t = threadIdx.x;

  float mx = -1e30f, s = 0.f;
  if (t == 0) {    // fold the 4 memory-kv tokens into thread 0's running pair
    for (int m = 0; m < 4; m++) {
      float v = memk[m];
      float nm = fmaxf(mx, v);
      s = s * __expf(mx - nm) + __expf(v - nm);
      mx = nm;
    }
  }
  for (int i = t; i < NTOT / 2; i += 256) {
    H2 p; p.u = row32[i];
    float v0 = (float)p.h[0], v1 = (float)p.h[1];
    float nm = fmaxf(mx, fmaxf(v0, v1));
    s = s * __expf(mx - nm) + __expf(v0 - nm) + __expf(v1 - nm);
    mx = nm;
  }
  red_m[t] = mx; red_s[t] = s; __syncthreads();
  for (int st = 128; st > 0; st >>= 1) {
    if (t < st) {
      float m1 = red_m[t], s1 = red_s[t];
      float m2 = red_m[t + st], s2 = red_s[t + st];
      float M = fmaxf(m1, m2);
      red_s[t] = s1 * __expf(m1 - M) + s2 * __expf(m2 - M);
      red_m[t] = M;
    }
    __syncthreads();
  }
  if (t == 0) { kmax[idx] = red_m[0]; ksum[idx] = red_s[0]; }
}

// ------- Kernel 5a: partial context over one n-chunk (deterministic partials) --
// ctxPart[cz, b, h, d, e] = sum_{n in chunk} exp(k[d,n]-max[d]) * v[e,n]
__global__ __launch_bounds__(256) void la_context_kernel(
    const _Float16* __restrict__ qkv, const float* __restrict__ kmax,
    float* __restrict__ ctxPart) {
  __shared__ float lk[32 * 128];
  __shared__ float lv[32 * 128];
  __shared__ float mxs[32];
  const int h = blockIdx.x, b = blockIdx.y, cz = blockIdx.z, t = threadIdx.x;
  if (t < 32) mxs[t] = kmax[b * 256 + h * 32 + t];
  __syncthreads();
  const int d = t >> 3;
  const int e0 = (t & 7) * 4;
  const int nBeg = cz * (NTOT / NCHUNK);
  const int nEnd = nBeg + (NTOT / NCHUNK);
  float acc[4] = {0.f, 0.f, 0.f, 0.f};
  const _Float16* kb = qkv + ((size_t)(b * QKVROWS + 256 + h * 32)) * NTOT;
  const _Float16* vb = qkv + ((size_t)(b * QKVROWS + 512 + h * 32)) * NTOT;
  for (int n0 = nBeg; n0 < nEnd; n0 += 128) {
#pragma unroll
    for (int i = 0; i < 8; i++) {
      int pidx = i * 256 + t;                 // 0..2047 f16-pairs
      int rr = pidx >> 6;                     // 0..31
      int c2 = pidx & 63;                     // pair index in row
      H2 pk; pk.u = ((const unsigned int*)(kb + (size_t)rr * NTOT + n0))[c2];
      H2 pv; pv.u = ((const unsigned int*)(vb + (size_t)rr * NTOT + n0))[c2];
      lk[rr * 128 + 2 * c2]     = __expf((float)pk.h[0] - mxs[rr]);
      lk[rr * 128 + 2 * c2 + 1] = __expf((float)pk.h[1] - mxs[rr]);
      lv[rr * 128 + 2 * c2]     = (float)pv.h[0];
      lv[rr * 128 + 2 * c2 + 1] = (float)pv.h[1];
    }
    // prefetch next tile (32 rows x 256B x {k,v}) -> 256 threads, 64B granules
    if (n0 + 128 < nEnd) {
      int rr = t & 31;
      const _Float16* pb = ((t & 32) ? vb : kb) + (size_t)rr * NTOT + n0 + 128;
      __builtin_prefetch(pb + ((t >> 6) & 3) * 32, 0, 3);
    }
    __syncthreads();
#pragma unroll 4
    for (int j = 0; j < 128; j++) {
      float kvv = lk[d * 128 + j];
      acc[0] += kvv * lv[(e0 + 0) * 128 + j];
      acc[1] += kvv * lv[(e0 + 1) * 128 + j];
      acc[2] += kvv * lv[(e0 + 2) * 128 + j];
      acc[3] += kvv * lv[(e0 + 3) * 128 + j];
    }
    __syncthreads();
  }
  float* dst = ctxPart + (size_t)cz * (16 * 8 * 32 * 32) +
               ((size_t)((b * 8 + h) * 32 + d)) * 32 + e0;
#pragma unroll
  for (int i = 0; i < 4; i++) dst[i] = acc[i];
}

// ------- Kernel 5b: finalize context: fixed-order chunk sum + mem kv + 1/sum ---
__global__ __launch_bounds__(256) void la_ctx_final_kernel(
    const float* __restrict__ ctxPart, const float* __restrict__ memkv,
    const float* __restrict__ kmax, const float* __restrict__ ksum,
    float* __restrict__ ctx) {
  __shared__ float mxs[32], sms[32];
  const int h = blockIdx.x, b = blockIdx.y, t = threadIdx.x;
  if (t < 32) { mxs[t] = kmax[b * 256 + h * 32 + t]; sms[t] = ksum[b * 256 + h * 32 + t]; }
  __syncthreads();
  const int d = t >> 3;
  const int e0 = (t & 7) * 4;
  const size_t base = ((size_t)((b * 8 + h) * 32 + d)) * 32 + e0;
  float acc[4] = {0.f, 0.f, 0.f, 0.f};
#pragma unroll
  for (int cz = 0; cz < NCHUNK; cz++) {
    const float* src = ctxPart + (size_t)cz * (16 * 8 * 32 * 32) + base;
#pragma unroll
    for (int i = 0; i < 4; i++) acc[i] += src[i];
  }
  const float* memk = memkv + (h * 32 + d) * 4;
  const float* memv = memkv + 1024;     // mem_kv[1]
  float inv = 1.f / sms[d];
#pragma unroll
  for (int i = 0; i < 4; i++) {
    float a = acc[i];
    for (int m = 0; m < 4; m++)
      a += __expf(memk[m] - mxs[d]) * memv[(h * 32 + e0 + i) * 4 + m];
    ctx[base + i] = a * inv;
  }
}

// ---------------- Kernel 6: fused q-softmax -> ctx.q -> w_out WMMA -> LN -> +xn
__global__ __launch_bounds__(256) void la_final_kernel(
    const _Float16* __restrict__ qkv, const float* __restrict__ ctx,
    const _Float16* __restrict__ wout, const float* __restrict__ b_out,
    const float* __restrict__ ln_g, const float* __restrict__ x,
    const float* __restrict__ meanX, const float* __restrict__ rstdX,
    float* __restrict__ out) {
  __shared__ char smem[65536];
  // region A (0..32KB): qs f16 (phase 1-2) -> wave partials + col stats (phase 4)
  // region B (32..64KB): attn f16 (phase 2-3) -> ldsY f32 staging (phase 5)
  _Float16* qs   = (_Float16*)smem;
  _Float16* attn = (_Float16*)(smem + 32768);
  float* wsum    = (float*)smem;                 // [16][64] per (wave,khalf) partial
  float* wsq     = (float*)(smem + 4096);        // [16][64]
  float* colMean = (float*)(smem + 8192);        // [64]
  float* colRstd = (float*)(smem + 8448);        // [64]
  float* ldsY    = (float*)(smem + 32768);       // [128][64] staging

  const int t = threadIdx.x;
  const int b = blockIdx.y;
  const int n0 = blockIdx.x * 64;

  // Phase 1: q softmax over d (32) per (h, n); fold SCALE=0.5
#pragma unroll
  for (int rep = 0; rep < 2; rep++) {
    int p = rep * 256 + t;       // 0..511
    int h = p >> 6, n = p & 63;
    const _Float16* qbase = qkv + ((size_t)(b * QKVROWS + h * 32)) * NTOT + n0 + n;
    float qv[32];
    float mx = -1e30f;
#pragma unroll
    for (int d = 0; d < 32; d++) { qv[d] = (float)qbase[(size_t)d * NTOT]; mx = fmaxf(mx, qv[d]); }
    float s = 0.f;
#pragma unroll
    for (int d = 0; d < 32; d++) { qv[d] = __expf(qv[d] - mx); s += qv[d]; }
    float inv = 0.5f / s;
#pragma unroll
    for (int d = 0; d < 32; d++) qs[(h * 32 + d) * 64 + n] = (_Float16)(qv[d] * inv);
  }
  __syncthreads();

  // Phase 2: attn[e,n] = sum_d ctx[d,e] * qs[d,n] per head (thread t = row e)
  {
    const int o = t, h = o >> 5, e = o & 31;
    float c[32];
    const float* cb = ctx + ((size_t)((b * 8 + h) * 32)) * 32 + e;
#pragma unroll
    for (int d = 0; d < 32; d++) c[d] = cb[d * 32];
    for (int n = 0; n < 64; n++) {
      float a = 0.f;
#pragma unroll
      for (int d = 0; d < 32; d++) a += c[d] * (float)qs[(h * 32 + d) * 64 + n];
      attn[o * 64 + n] = (_Float16)a;
    }
  }
  __syncthreads();

  // Phase 3: y[256x64] = w_out(f16) @ attn(f16), f32 accum. Wave w -> rows [w*32, w*32+32)
  const int lane = t & 31;
  const int w = t >> 5;
  const int l15 = lane & 15;
  const int khalf = lane >> 4;
  v8f acc[2][4];
#pragma unroll
  for (int i = 0; i < 2; i++)
#pragma unroll
    for (int j = 0; j < 4; j++) acc[i][j] = {};
  for (int k0 = 0; k0 < CDIM; k0 += 32) {
    AFrag a[2];
#pragma unroll
    for (int mi = 0; mi < 2; mi++) {
      int m = w * 32 + mi * 16 + l15;
      const _Float16* wr = wout + (size_t)m * CDIM + k0 + khalf * 16;
      a[mi].q2[0] = *(const u32x4*)wr;
      a[mi].q2[1] = *(const u32x4*)(wr + 8);
    }
    BFrag bf[4];
#pragma unroll
    for (int nj = 0; nj < 4; nj++) {
      int n = nj * 16 + l15;
      int kb = k0 + khalf * 16;
#pragma unroll
      for (int j = 0; j < 8; j++) {
        H2 p;
        p.h[0] = attn[(kb + 2 * j)     * 64 + n];
        p.h[1] = attn[(kb + 2 * j + 1) * 64 + n];
        bf[nj].u[j] = p.u;
      }
    }
#pragma unroll
    for (int mi = 0; mi < 2; mi++)
#pragma unroll
      for (int nj = 0; nj < 4; nj++)
        acc[mi][nj] = wmma_f16(a[mi].v, bf[nj].v, acc[mi][nj]);
  }

  // Phase 4: per-column mean/var over 256 channels, deterministic fixed-order tree
  float bias[2][8];
#pragma unroll
  for (int mi = 0; mi < 2; mi++)
#pragma unroll
    for (int v = 0; v < 8; v++)
      bias[mi][v] = b_out[w * 32 + mi * 16 + (khalf ? 8 : 0) + v];
#pragma unroll
  for (int nj = 0; nj < 4; nj++) {
    float ls = 0.f, lq = 0.f;
#pragma unroll
    for (int mi = 0; mi < 2; mi++)
#pragma unroll
      for (int v = 0; v < 8; v++) {
        float y = acc[mi][nj][v] + bias[mi][v];
        ls += y; lq += y * y;
      }
    int slot = (w * 2 + khalf) * 64 + nj * 16 + l15;   // region A is free now
    wsum[slot] = ls;
    wsq[slot] = lq;
  }
  __syncthreads();   // also: all waves done reading attn (region B reusable)
  if (t < 64) {
    float s = 0.f, q = 0.f;
#pragma unroll
    for (int i = 0; i < 16; i++) { s += wsum[i * 64 + t]; q += wsq[i * 64 + t]; }
    float mu = s / 256.f;
    float var = q / 256.f - mu * mu;
    colMean[t] = mu;
    colRstd[t] = rsqrtf(var + EPSV);
  }
  __syncthreads();

  // Phase 5: LN * g, stage 128-row halves in LDS, then coalesced b128 resid+store
  for (int r = 0; r < 2; r++) {
    if ((w >> 2) == r) {
#pragma unroll
      for (int mi = 0; mi < 2; mi++)
#pragma unroll
        for (int nj = 0; nj < 4; nj++) {
          int nn = nj * 16 + l15;
#pragma unroll
          for (int v = 0; v < 8; v++) {
            int m = w * 32 + mi * 16 + (khalf ? 8 : 0) + v;
            float y = acc[mi][nj][v] + bias[mi][v];
            y = (y - colMean[nn]) * colRstd[nn] * ln_g[m];
            ldsY[(m - r * 128) * 64 + nn] = y;
          }
        }
    }
    __syncthreads();
    {
      int row = t >> 1;                  // 0..127
      int m = r * 128 + row;
      int c0 = (t & 1) * 8;              // float4 base within the 16-float4 row
      float mu = meanX[b * CDIM + m], rs = rstdX[b * CDIM + m];
      const f32x4* xr = (const f32x4*)(x + ((size_t)(b * CDIM + m)) * NTOT + n0);
      f32x4* orow = (f32x4*)(out + ((size_t)(b * CDIM + m)) * NTOT + n0);
      const f32x4* yr = (const f32x4*)(ldsY + row * 64);
#pragma unroll
      for (int i = 0; i < 8; i++) {
        f32x4 xv = xr[c0 + i];
        f32x4 yv = yr[c0 + i];
        f32x4 o;
        o.x = yv.x + (xv.x - mu) * rs;
        o.y = yv.y + (xv.y - mu) * rs;
        o.z = yv.z + (xv.z - mu) * rs;
        o.w = yv.w + (xv.w - mu) * rs;
        orow[c0 + i] = o;
      }
    }
    __syncthreads();
  }
}

// ---------------- Host launcher ----------------
extern "C" void kernel_launch(void* const* d_in, const int* in_sizes, int n_in,
                              void* d_out, int out_size, void* d_ws, size_t ws_size,
                              hipStream_t stream) {
  const float* x      = (const float*)d_in[0];   // [16,256,128,128]
  const float* w_qkv  = (const float*)d_in[1];   // [768,256]
  const float* mem_kv = (const float*)d_in[2];   // [2,8,32,4]
  const float* w_out  = (const float*)d_in[3];   // [256,256]
  const float* b_out  = (const float*)d_in[4];   // [256]
  const float* ln_g   = (const float*)d_in[5];   // [256]
  float* out = (float*)d_out;

  // workspace layout (16B aligned)
  char* ws = (char*)d_ws;
  _Float16* qkv16  = (_Float16*)(ws);                      // 16*768*16384 f16 = 384 MB
  size_t off = (size_t)16 * QKVROWS * NTOT * sizeof(_Float16);
  float*    kmax    = (float*)(ws + off); off += 4096 * sizeof(float);
  float*    ksum    = (float*)(ws + off); off += 4096 * sizeof(float);
  float*    ctx     = (float*)(ws + off); off += (size_t)16 * 8 * 32 * 32 * sizeof(float);
  float*    ctxPart = (float*)(ws + off); off += (size_t)NCHUNK * 16 * 8 * 32 * 32 * sizeof(float);
  _Float16* wq16    = (_Float16*)(ws + off); off += (size_t)QKVROWS * CDIM * sizeof(_Float16);
  _Float16* wout16  = (_Float16*)(ws + off); off += (size_t)CDIM * CDIM * sizeof(_Float16);
  float*    meanX   = (float*)(ws + off); off += 4096 * sizeof(float);
  float*    rstdX   = (float*)(ws + off); off += 4096 * sizeof(float);
  (void)ws_size; (void)in_sizes; (void)n_in; (void)out_size;

  la_stats_kernel<<<dim3(4096), 256, 0, stream>>>(x, meanX, rstdX);
  la_cvt_kernel<<<dim3((QKVROWS * CDIM + 255) / 256), 256, 0, stream>>>(w_qkv, wq16, QKVROWS * CDIM);
  la_cvt_kernel<<<dim3((CDIM * CDIM + 255) / 256), 256, 0, stream>>>(w_out, wout16, CDIM * CDIM);
  la_gemm_qkv_kernel<<<dim3(128, 6, 16), 256, 0, stream>>>(x, meanX, rstdX, wq16, qkv16);
  la_kstats_kernel<<<dim3(4096), 256, 0, stream>>>(qkv16, mem_kv, kmax, ksum);
  la_context_kernel<<<dim3(8, 16, NCHUNK), 256, 0, stream>>>(qkv16, kmax, ctxPart);
  la_ctx_final_kernel<<<dim3(8, 16), 256, 0, stream>>>(ctxPart, mem_kv, kmax, ksum, ctx);
  la_final_kernel<<<dim3(256, 16), 256, 0, stream>>>(qkv16, ctx, wout16, b_out, ln_g,
                                                     x, meanX, rstdX, out);
}
